// CosineSimCodebook_8804682957471
// MI455X (gfx1250) — matrix operand order repaired and checked by reading, not compiled, not run on hardware
//
#include <hip/hip_runtime.h>
#include <stdint.h>

// Problem constants (from reference shapes)
#define DDIM   512
#define NROWS  16384   // 16 * 1024 flattened tokens
#define CBOOK  8192    // codebook size

typedef __attribute__((ext_vector_type(8)))  __bf16 v8bf;
typedef __attribute__((ext_vector_type(16))) __bf16 v16bf;
typedef __attribute__((ext_vector_type(8)))  float  v8f;

// ---- fragment loaders -------------------------------------------------------
// A frag (16x32 bf16, per ISA): lane holds K = kb+8*half+{0..7} and kb+16+8*half+{0..7}
__device__ __forceinline__ v16bf ld_frag_a(const __bf16* p) {
  v8bf a = *(const v8bf*)p;          // 16B
  v8bf b = *(const v8bf*)(p + 16);   // 16B, gap of 8 elems
  return __builtin_shufflevector(a, b, 0,1,2,3,4,5,6,7,8,9,10,11,12,13,14,15);
}
// B frag (32x16 bf16): lane holds K = kb+8*half+{0..7} and kb+16+8*half+{0..7}
__device__ __forceinline__ v16bf ld_frag_b(const __bf16* p) {
  v8bf a = *(const v8bf*)p;
  v8bf b = *(const v8bf*)(p + 16);
  return __builtin_shufflevector(a, b, 0,1,2,3,4,5,6,7,8,9,10,11,12,13,14,15);
}

// Monotone float encoding, high word = value, low word = ~index (ties -> lowest index)
__device__ __forceinline__ unsigned long long make_key(float f, unsigned c) {
  unsigned u = __float_as_uint(f);
  u = (u & 0x80000000u) ? ~u : (u | 0x80000000u);
  return ((unsigned long long)u << 32) | (0xFFFFFFFFu - c);
}
__device__ __forceinline__ unsigned long long shfl_xor_u64(unsigned long long v, int m) {
  unsigned lo = (unsigned)v, hi = (unsigned)(v >> 32);
  lo = __shfl_xor(lo, m, 32);
  hi = __shfl_xor(hi, m, 32);
  return ((unsigned long long)hi << 32) | lo;
}

// ---- kernel 0: zero argmax keys --------------------------------------------
__global__ void vq_init_keys(unsigned long long* __restrict__ keys) {
  int i = blockIdx.x * blockDim.x + threadIdx.x;
  if (i < NROWS) keys[i] = 0ull;
}

// ---- kernel 1: l2norm + bf16 hi/lo split -----------------------------------
// One block per row of 512. dstN (optional) gets fp32 normalized row.
__global__ void __launch_bounds__(256)
vq_norm_split(const float* __restrict__ src, float* __restrict__ dstN,
              __bf16* __restrict__ dstHi, __bf16* __restrict__ dstLo) {
  __shared__ float red[8];
  const int row = blockIdx.x;
  const int t = threadIdx.x;
  const size_t base = (size_t)row * DDIM;
  float f0 = src[base + t];
  float f1 = src[base + t + 256];
  float s = f0 * f0 + f1 * f1;
  #pragma unroll
  for (int m = 16; m >= 1; m >>= 1) s += __shfl_xor(s, m, 32);
  if ((t & 31) == 0) red[t >> 5] = s;
  __syncthreads();
  float tot = red[0] + red[1] + red[2] + red[3] + red[4] + red[5] + red[6] + red[7];
  float inv = 1.0f / fmaxf(sqrtf(tot), 1e-12f);
  float n0 = f0 * inv, n1 = f1 * inv;
  if (dstN) { dstN[base + t] = n0; dstN[base + t + 256] = n1; }
  __bf16 h0 = (__bf16)n0, h1 = (__bf16)n1;
  dstHi[base + t]       = h0;
  dstHi[base + t + 256] = h1;
  dstLo[base + t]       = (__bf16)(n0 - (float)h0);
  dstLo[base + t + 256] = (__bf16)(n1 - (float)h1);
}

// ---- kernel 2: dist GEMM (3x-bf16 compensated) + fused argmax --------------
// Block: 256 threads = 8 waves (4 along M, 2 along N). Block tile 128x64.
// Wave tile 32x32 = 2x2 frags of 16x16; K loop in chunks of 32.
__global__ void __launch_bounds__(256)
vq_dist_wmma(const __bf16* __restrict__ Ahi, const __bf16* __restrict__ Alo,
             const __bf16* __restrict__ Bhi, const __bf16* __restrict__ Blo,
             float* __restrict__ dist, unsigned long long* __restrict__ keys) {
  const int lane  = threadIdx.x & 31;
  const int wave  = threadIdx.x >> 5;
  const int waveM = wave & 3;
  const int waveN = wave >> 2;
  const int rowBase = blockIdx.y * 128 + waveM * 32;
  const int colBase = blockIdx.x * 64 + waveN * 32;
  const int r  = lane & 15;
  const int hf = lane >> 4;

  v8f acc[2][2] = {{{}, {}}, {{}, {}}};

  const size_t aRow0 = (size_t)(rowBase + r) * DDIM + 8 * hf;
  const size_t aRow1 = aRow0 + (size_t)16 * DDIM;
  const size_t bRow0 = (size_t)(colBase + r) * DDIM + 8 * hf;
  const size_t bRow1 = bRow0 + (size_t)16 * DDIM;

  for (int kb = 0; kb < DDIM; kb += 32) {
    v16bf a0h = ld_frag_a(Ahi + aRow0 + kb);
    v16bf a1h = ld_frag_a(Ahi + aRow1 + kb);
    v16bf a0l = ld_frag_a(Alo + aRow0 + kb);
    v16bf a1l = ld_frag_a(Alo + aRow1 + kb);
    v16bf b0h = ld_frag_b(Bhi + bRow0 + kb);
    v16bf b1h = ld_frag_b(Bhi + bRow1 + kb);
    v16bf b0l = ld_frag_b(Blo + bRow0 + kb);
    v16bf b1l = ld_frag_b(Blo + bRow1 + kb);

    // hi*hi + hi*lo + lo*hi  (lo*lo term ~1.6e-5 relative, dropped)
    acc[0][0] = __builtin_amdgcn_wmma_f32_16x16x32_bf16(false, a0h, false, b0h, (short)0, acc[0][0], false, false);
    acc[0][0] = __builtin_amdgcn_wmma_f32_16x16x32_bf16(false, a0h, false, b0l, (short)0, acc[0][0], false, false);
    acc[0][0] = __builtin_amdgcn_wmma_f32_16x16x32_bf16(false, a0l, false, b0h, (short)0, acc[0][0], false, false);

    acc[0][1] = __builtin_amdgcn_wmma_f32_16x16x32_bf16(false, a0h, false, b1h, (short)0, acc[0][1], false, false);
    acc[0][1] = __builtin_amdgcn_wmma_f32_16x16x32_bf16(false, a0h, false, b1l, (short)0, acc[0][1], false, false);
    acc[0][1] = __builtin_amdgcn_wmma_f32_16x16x32_bf16(false, a0l, false, b1h, (short)0, acc[0][1], false, false);

    acc[1][0] = __builtin_amdgcn_wmma_f32_16x16x32_bf16(false, a1h, false, b0h, (short)0, acc[1][0], false, false);
    acc[1][0] = __builtin_amdgcn_wmma_f32_16x16x32_bf16(false, a1h, false, b0l, (short)0, acc[1][0], false, false);
    acc[1][0] = __builtin_amdgcn_wmma_f32_16x16x32_bf16(false, a1l, false, b0h, (short)0, acc[1][0], false, false);

    acc[1][1] = __builtin_amdgcn_wmma_f32_16x16x32_bf16(false, a1h, false, b1h, (short)0, acc[1][1], false, false);
    acc[1][1] = __builtin_amdgcn_wmma_f32_16x16x32_bf16(false, a1h, false, b1l, (short)0, acc[1][1], false, false);
    acc[1][1] = __builtin_amdgcn_wmma_f32_16x16x32_bf16(false, a1l, false, b1h, (short)0, acc[1][1], false, false);
  }

  // C/D layout: VGPR v, lanes 0-15 -> M = v (N = lane); lanes 16-31 -> M = 8 + v.
  // Stream dist with non-temporal stores (512 MB, keep L2 for A/B reuse).
  #pragma unroll
  for (int mf = 0; mf < 2; ++mf) {
    #pragma unroll
    for (int nf = 0; nf < 2; ++nf) {
      const int cg = colBase + 16 * nf + r;
      #pragma unroll
      for (int v = 0; v < 8; ++v) {
        const int rowg = rowBase + 16 * mf + 8 * hf + v;
        __builtin_nontemporal_store(acc[mf][nf][v], dist + (size_t)rowg * CBOOK + cg);
      }
    }
  }

  // Fused per-row argmax over this block's 64-column slab -> device atomic max.
  #pragma unroll
  for (int mf = 0; mf < 2; ++mf) {
    #pragma unroll
    for (int v = 0; v < 8; ++v) {
      const int rowg = rowBase + 16 * mf + 8 * hf + v;
      unsigned long long k0 = make_key(acc[mf][0][v], (unsigned)(colBase + r));
      unsigned long long k1 = make_key(acc[mf][1][v], (unsigned)(colBase + 16 + r));
      unsigned long long k = (k0 > k1) ? k0 : k1;
      #pragma unroll
      for (int m = 1; m <= 8; m <<= 1) {   // reduce across the 16 lanes of this half
        unsigned long long o = shfl_xor_u64(k, m);
        if (o > k) k = o;
      }
      if (r == 0) atomicMax(keys + rowg, k);
    }
  }
}

// ---- kernel 3: decode index, write embed_ind, gather quantize --------------
__global__ void __launch_bounds__(128)
vq_finalize(const unsigned long long* __restrict__ keys,
            const float* __restrict__ embedN,
            float* __restrict__ quant, float* __restrict__ ind) {
  const int n = blockIdx.x;
  const unsigned long long k = keys[n];
  const unsigned idx = 0xFFFFFFFFu - (unsigned)(k & 0xFFFFFFFFull);
  if (threadIdx.x == 0) ind[n] = (float)idx;
  const float4* src = (const float4*)(embedN + (size_t)idx * DDIM);
  float4* dst = (float4*)(quant + (size_t)n * DDIM);
  dst[threadIdx.x] = src[threadIdx.x];   // 128 threads x float4 = 512 floats
}

extern "C" void kernel_launch(void* const* d_in, const int* in_sizes, int n_in,
                              void* d_out, int out_size, void* d_ws, size_t ws_size,
                              hipStream_t stream) {
  (void)in_sizes; (void)n_in; (void)out_size; (void)ws_size;
  const float* x     = (const float*)d_in[0];   // [1,16,1024,512]
  const float* embed = (const float*)d_in[1];   // [1,8192,512]

  float* out   = (float*)d_out;
  float* quant = out;                                   // 16384*512
  float* ind   = out + (size_t)NROWS * DDIM;            // 16384
  float* dist  = ind + NROWS;                           // 16384*8192

  // Workspace: embed_n fp32 (16MB) + E_hi (8MB) + E_lo (8MB) + keys (128KB)
  char* ws = (char*)d_ws;
  float*  embedN = (float*)ws;
  __bf16* Ehi = (__bf16*)(ws + (size_t)CBOOK * DDIM * 4);
  __bf16* Elo = (__bf16*)((char*)Ehi + (size_t)CBOOK * DDIM * 2);
  unsigned long long* keys = (unsigned long long*)((char*)Elo + (size_t)CBOOK * DDIM * 2);

  // A_hi/A_lo (16MB each) live in the quantize region of d_out (exactly 32MB);
  // the region is fully rewritten by vq_finalize after the GEMM consumes them.
  __bf16* Ahi = (__bf16*)quant;
  __bf16* Alo = (__bf16*)((char*)quant + (size_t)NROWS * DDIM * 2);

  vq_init_keys<<<NROWS / 256, 256, 0, stream>>>(keys);
  vq_norm_split<<<NROWS, 256, 0, stream>>>(x, nullptr, Ahi, Alo);
  vq_norm_split<<<CBOOK, 256, 0, stream>>>(embed, embedN, Ehi, Elo);

  dim3 grid(CBOOK / 64, NROWS / 128);   // 128 x 128 blocks
  vq_dist_wmma<<<grid, 256, 0, stream>>>(Ahi, Alo, Ehi, Elo, dist, keys);

  vq_finalize<<<NROWS, 128, 0, stream>>>(keys, embedN, quant, ind);
}
